// GGNNSumMulCategory_26405458935923
// MI455X (gfx1250) — matrix operand order repaired and checked by reading, not compiled, not run on hardware
//
#include <hip/hip_runtime.h>

#define NODES   50000
#define EDGES   320000
#define INDIM   128
#define OUTDIM  256
#define ETYPES  4
#define STEPS   8
#define GRAPHS  256
#define CLASSES 7

typedef __bf16 bf16_t;
typedef bf16_t v8bf  __attribute__((ext_vector_type(8)));
typedef bf16_t v16bf __attribute__((ext_vector_type(16)));
typedef float  v8f   __attribute__((ext_vector_type(8)));

static __device__ __forceinline__ unsigned short f2bf(float f) {
  unsigned int u = __float_as_uint(f);
  u += 0x7FFFu + ((u >> 16) & 1u);   // round-to-nearest-even
  return (unsigned short)(u >> 16);
}

static __device__ __forceinline__ v16bf cat8(v8bf lo, v8bf hi) {
  return __builtin_shufflevector(lo, hi, 0,1,2,3,4,5,6,7,8,9,10,11,12,13,14,15);
}

// async-copy 32KB (contiguous) global -> LDS, 256 threads x 8 x 16B
static __device__ __forceinline__ void async_copy_panel32k(const unsigned short* gsrc,
                                                           unsigned int ldsBase, int tid) {
  unsigned long long g = (unsigned long long)gsrc + (unsigned long long)tid * 16;
  unsigned int l = ldsBase + (unsigned int)tid * 16;
#pragma unroll
  for (int r = 0; r < 8; ++r) {
    asm volatile("global_load_async_to_lds_b128 %0, %1, off"
                 :: "v"(l + r * 4096u), "v"(g + (unsigned long long)(r * 4096)) : "memory");
  }
  asm volatile("s_wait_asynccnt 0x0" ::: "memory");
}

// ---------------- utility kernels ----------------

__global__ void zero_u32_kernel(unsigned int* p, long long n) {
  long long i = (long long)blockIdx.x * blockDim.x + threadIdx.x;
  long long s = (long long)gridDim.x * blockDim.x;
  for (; i < n; i += s) p[i] = 0u;
}

__global__ void f32_to_bf16_kernel(const float* __restrict__ src,
                                   unsigned short* __restrict__ dst, long long n) {
  long long i = (long long)blockIdx.x * blockDim.x + threadIdx.x;
  long long s = (long long)gridDim.x * blockDim.x;
  for (; i < n; i += s) dst[i] = f2bf(src[i]);
}

// h = pad(x, 128 -> 256); hb = bf16(h).   grid: NODES blocks x 256 threads
__global__ void init_h_kernel(const float* __restrict__ x,
                              float* __restrict__ h, unsigned short* __restrict__ hb) {
  int v = blockIdx.x, j = threadIdx.x;
  float val = (j < INDIM) ? x[(size_t)v * INDIM + j] : 0.0f;
  size_t i = (size_t)v * OUTDIM + j;
  h[i] = val;
  hb[i] = f2bf(val);
}

// ---------------- edge scatter: s[t][dst] += hb[src] (packed bf16 atomics) ----------------
__global__ void edge_scatter_kernel(const int* __restrict__ src, const int* __restrict__ dst,
                                    const int* __restrict__ et,
                                    const unsigned short* __restrict__ hb,
                                    unsigned short* __restrict__ s,
                                    float* __restrict__ cnt) {
  int wave = (int)((blockIdx.x * blockDim.x + threadIdx.x) >> 5);
  int lane = (int)(threadIdx.x & 31);
  if (wave >= EDGES) return;
  int u = src[wave], v = dst[wave], t = et[wave];
  uint4 d = *(const uint4*)(hb + (size_t)u * OUTDIM + lane * 8);
  unsigned long long base =
      (unsigned long long)(s + ((size_t)t * NODES + v) * OUTDIM + lane * 8);
  asm volatile("global_atomic_pk_add_bf16 %0, %1, off" :: "v"(base),      "v"(d.x) : "memory");
  asm volatile("global_atomic_pk_add_bf16 %0, %1, off" :: "v"(base + 4),  "v"(d.y) : "memory");
  asm volatile("global_atomic_pk_add_bf16 %0, %1, off" :: "v"(base + 8),  "v"(d.z) : "memory");
  asm volatile("global_atomic_pk_add_bf16 %0, %1, off" :: "v"(base + 12), "v"(d.w) : "memory");
  if (lane == 0) unsafeAtomicAdd(cnt + (size_t)t * NODES + v, 1.0f);
}

// ---------------- WMMA GEMM core: one wave, 16x64 tile, B from LDS panel ----------------
// panel: 64 rows x 256 cols bf16 (row = output column n0+r).  A: bf16 row-major lda=256.
static __device__ __forceinline__ void wave_tile_gemm(const unsigned short* __restrict__ Arow,
                                                      const unsigned short* __restrict__ panel,
                                                      int half, int l16, v8f c[4]) {
  v16bf af = cat8(*(const v8bf*)(Arow + half * 8),
                  *(const v8bf*)(Arow + 16 + half * 8));
#pragma unroll
  for (int k = 0; k < OUTDIM; k += 32) {
    v16bf afn = af;
    if (k + 32 < OUTDIM)  // prefetch next slab while this slab's WMMAs run
      afn = cat8(*(const v8bf*)(Arow + k + 32 + half * 8),
                 *(const v8bf*)(Arow + k + 48 + half * 8));
#pragma unroll
    for (int j = 0; j < 4; ++j) {
      const unsigned short* Bp = panel + (size_t)(j * 16 + l16) * OUTDIM + k + half * 8;
      v16bf bm = cat8(*(const v8bf*)Bp, *(const v8bf*)(Bp + 16));
      c[j] = __builtin_amdgcn_wmma_f32_16x16x32_bf16(false, af, false, bm,
                                                     (short)0, c[j], false, false);
    }
    af = afn;
  }
}

// ---------------- message GEMM: a = sum_t s_t @ W_t^T  (K = 4*256) ----------------
// block = 8 waves sharing one 64-col strip; W_t panel staged in LDS per etype
__global__ void __launch_bounds__(256) msg_gemm_kernel(
    const unsigned short* __restrict__ S,   // [4, NODES, 256] bf16
    const unsigned short* __restrict__ Wm,  // [4, 256, 256] bf16
    float* __restrict__ outA, int Mtiles) {
  __shared__ __align__(16) unsigned short panel[64 * OUTDIM];  // 32 KB
  const int tid = threadIdx.x;
  const int nstrip = blockIdx.x & 3;
  const int mblock = blockIdx.x >> 2;
  const int n0 = nstrip * 64;
  const int wave = tid >> 5, lane = tid & 31;
  const int half = lane >> 4, l16 = lane & 15;
  const int mt = mblock * 8 + wave;
  const bool active = (mt < Mtiles);          // wave-uniform => EXEC stays all-ones
  const int m0 = mt * 16;
  const unsigned int ldsBase = (unsigned int)(size_t)&panel[0];
  v8f c[4] = {};
  for (int t = 0; t < ETYPES; ++t) {
    async_copy_panel32k(Wm + ((size_t)t * OUTDIM + n0) * OUTDIM, ldsBase, tid);
    __syncthreads();
    if (active) {
      const unsigned short* Arow = S + ((size_t)t * NODES + (m0 + l16)) * OUTDIM;
      wave_tile_gemm(Arow, panel, half, l16, c);
    }
    __syncthreads();
  }
  if (active) {
#pragma unroll
    for (int j = 0; j < 4; ++j) {
      int ncol = n0 + j * 16 + l16;
#pragma unroll
      for (int i = 0; i < 8; ++i)
        outA[(size_t)(m0 + half * 8 + i) * OUTDIM + ncol] = c[j][i];
    }
  }
}

// ---------------- generic GEMM: out[M,*] = A(bf16)[M,256] @ W(bf16)[*,256]^T + bias ----------------
__global__ void __launch_bounds__(256) gemm_wmma_kernel(
    const unsigned short* __restrict__ A,
    const unsigned short* __restrict__ W,
    const float* __restrict__ bias,
    float* __restrict__ out, int ldo,
    int Mtiles, int Nstrips) {
  __shared__ __align__(16) unsigned short panel[64 * OUTDIM];  // 32 KB
  const int tid = threadIdx.x;
  const int nstrip = blockIdx.x % Nstrips;
  const int mblock = blockIdx.x / Nstrips;
  const int n0 = nstrip * 64;
  async_copy_panel32k(W + (size_t)n0 * OUTDIM, (unsigned int)(size_t)&panel[0], tid);
  __syncthreads();

  const int wave = tid >> 5, lane = tid & 31;
  const int half = lane >> 4, l16 = lane & 15;
  const int mt = mblock * 8 + wave;
  if (mt >= Mtiles) return;                  // after last barrier; wave-uniform
  const int m0 = mt * 16;
  v8f c[4] = {};
  wave_tile_gemm(A + (size_t)(m0 + l16) * OUTDIM, panel, half, l16, c);
#pragma unroll
  for (int j = 0; j < 4; ++j) {
    int ncol = n0 + j * 16 + l16;
    float bv = bias ? bias[ncol] : 0.0f;
#pragma unroll
    for (int i = 0; i < 8; ++i)
      out[(size_t)(m0 + half * 8 + i) * ldo + ncol] = c[j][i] + bv;
  }
}

// a += sum_t cnt[t][v] * b_msg[t]; ab = bf16(a).   grid: NODES x 256
__global__ void bias_convert_kernel(const float* __restrict__ a,
                                    const float* __restrict__ cnt,
                                    const float* __restrict__ b_msg,
                                    unsigned short* __restrict__ ab) {
  int v = blockIdx.x, j = threadIdx.x;
  float acc = a[(size_t)v * OUTDIM + j];
#pragma unroll
  for (int t = 0; t < ETYPES; ++t)
    acc += cnt[(size_t)t * NODES + v] * b_msg[t * OUTDIM + j];
  ab[(size_t)v * OUTDIM + j] = f2bf(acc);
}

// GRU elementwise.  grid: NODES x 256
__global__ void gru_kernel(const float* __restrict__ gi, const float* __restrict__ gh,
                           float* __restrict__ h, unsigned short* __restrict__ hb) {
  int v = blockIdx.x, j = threadIdx.x;
  const float* giv = gi + (size_t)v * 3 * OUTDIM;
  const float* ghv = gh + (size_t)v * 3 * OUTDIM;
  float r = 1.0f / (1.0f + __expf(-(giv[j] + ghv[j])));
  float z = 1.0f / (1.0f + __expf(-(giv[OUTDIM + j] + ghv[OUTDIM + j])));
  float n = tanhf(giv[2 * OUTDIM + j] + r * ghv[2 * OUTDIM + j]);
  size_t i = (size_t)v * OUTDIM + j;
  float hv = h[i];
  float hn = (1.0f - z) * n + z * hv;
  h[i] = hn;
  hb[i] = f2bf(hn);
}

// pooled[graph_ids[v]] += h[v].  grid: NODES x 256
__global__ void pool_kernel(const float* __restrict__ h, const int* __restrict__ gid,
                            float* __restrict__ pooled) {
  int v = blockIdx.x, j = threadIdx.x;
  unsafeAtomicAdd(&pooled[(size_t)gid[v] * OUTDIM + j], h[(size_t)v * OUTDIM + j]);
}

// logits + softmax.  1 block x 256 threads (one per graph)
__global__ void classify_kernel(const float* __restrict__ pooled,
                                const float* __restrict__ cw, const float* __restrict__ cb,
                                float* __restrict__ out) {
  int g = threadIdx.x;
  const float* pg = pooled + (size_t)g * OUTDIM;
  float lg[CLASSES];
  float mx = -1e30f;
  for (int c = 0; c < CLASSES; ++c) {
    float s = cb[c];
    for (int k = 0; k < OUTDIM; ++k) s += pg[k] * cw[c * OUTDIM + k];
    lg[c] = s;
    mx = fmaxf(mx, s);
  }
  float den = 0.0f;
  for (int c = 0; c < CLASSES; ++c) { lg[c] = __expf(lg[c] - mx); den += lg[c]; }
  float inv = 1.0f / den;
  for (int c = 0; c < CLASSES; ++c) out[(size_t)g * CLASSES + c] = lg[c] * inv;
}

// ---------------- host launch ----------------

extern "C" void kernel_launch(void* const* d_in, const int* in_sizes, int n_in,
                              void* d_out, int out_size, void* d_ws, size_t ws_size,
                              hipStream_t stream) {
  (void)in_sizes; (void)n_in; (void)out_size; (void)ws_size;
  const float* x     = (const float*)d_in[0];
  const int*   src   = (const int*)d_in[1];
  const int*   dst   = (const int*)d_in[2];
  const int*   et    = (const int*)d_in[3];
  const int*   gid   = (const int*)d_in[4];
  const float* W_msg = (const float*)d_in[5];
  const float* b_msg = (const float*)d_in[6];
  const float* w_ih  = (const float*)d_in[7];
  const float* w_hh  = (const float*)d_in[8];
  const float* b_ih  = (const float*)d_in[9];
  const float* b_hh  = (const float*)d_in[10];
  const float* cls_w = (const float*)d_in[11];
  const float* cls_b = (const float*)d_in[12];

  char* ws = (char*)d_ws;
  size_t off = 0;
  auto alloc = [&](size_t bytes) -> void* {
    void* p = ws + off;
    off = (off + bytes + 255) & ~(size_t)255;
    return p;
  };
  // total ~566 MB
  float*          h     = (float*)alloc((size_t)NODES * OUTDIM * 4);
  unsigned short* hb    = (unsigned short*)alloc((size_t)NODES * OUTDIM * 2);
  unsigned short* s     = (unsigned short*)alloc((size_t)ETYPES * NODES * OUTDIM * 2);
  float*          cnt   = (float*)alloc((size_t)ETYPES * NODES * 4);  // contiguous after s
  float*          a     = (float*)alloc((size_t)NODES * OUTDIM * 4);
  unsigned short* ab    = (unsigned short*)alloc((size_t)NODES * OUTDIM * 2);
  float*          gi    = (float*)alloc((size_t)NODES * 3 * OUTDIM * 4);
  float*          gh    = (float*)alloc((size_t)NODES * 3 * OUTDIM * 4);
  unsigned short* wmsgb = (unsigned short*)alloc((size_t)ETYPES * OUTDIM * OUTDIM * 2);
  unsigned short* wihb  = (unsigned short*)alloc((size_t)3 * OUTDIM * OUTDIM * 2);
  unsigned short* whhb  = (unsigned short*)alloc((size_t)3 * OUTDIM * OUTDIM * 2);
  float*          pooled= (float*)alloc((size_t)GRAPHS * OUTDIM * 4);

  // weight conversion to bf16
  f32_to_bf16_kernel<<<512, 256, 0, stream>>>(W_msg, wmsgb, (long long)ETYPES * OUTDIM * OUTDIM);
  f32_to_bf16_kernel<<<512, 256, 0, stream>>>(w_ih, wihb, (long long)3 * OUTDIM * OUTDIM);
  f32_to_bf16_kernel<<<512, 256, 0, stream>>>(w_hh, whhb, (long long)3 * OUTDIM * OUTDIM);
  init_h_kernel<<<NODES, 256, 0, stream>>>(x, h, hb);

  const int Mtiles  = NODES / 16;                 // 3125
  const int Mblocks = (Mtiles + 7) / 8;           // 391
  const long long zeroN = ((long long)ETYPES * NODES * OUTDIM * 2 +   // s bytes
                           (long long)ETYPES * NODES * 4) / 4;        // + cnt, as u32

  for (int step = 0; step < STEPS; ++step) {
    zero_u32_kernel<<<4096, 256, 0, stream>>>((unsigned int*)s, zeroN);
    edge_scatter_kernel<<<(EDGES * 32) / 256, 256, 0, stream>>>(src, dst, et, hb, s, cnt);
    msg_gemm_kernel<<<Mblocks * 4, 256, 0, stream>>>(s, wmsgb, a, Mtiles);
    bias_convert_kernel<<<NODES, 256, 0, stream>>>(a, cnt, b_msg, ab);
    gemm_wmma_kernel<<<Mblocks * 12, 256, 0, stream>>>(ab, wihb, b_ih, gi, 3 * OUTDIM, Mtiles, 12);
    gemm_wmma_kernel<<<Mblocks * 12, 256, 0, stream>>>(hb, whhb, b_hh, gh, 3 * OUTDIM, Mtiles, 12);
    gru_kernel<<<NODES, 256, 0, stream>>>(gi, gh, h, hb);
  }

  zero_u32_kernel<<<256, 256, 0, stream>>>((unsigned int*)pooled, (long long)GRAPHS * OUTDIM);
  pool_kernel<<<NODES, 256, 0, stream>>>(h, gid, pooled);
  classify_kernel<<<1, 256, 0, stream>>>(pooled, cls_w, cls_b, (float*)d_out);
}
